// custom_rnn_20323785244792
// MI455X (gfx1250) — matrix operand order repaired and checked by reading, not compiled, not run on hardware
//
#include <hip/hip_runtime.h>
#include <stdint.h>

#define SEQ 4096
#define HID 2048
#define NL 4

typedef __attribute__((ext_vector_type(16))) __bf16 v16bf;
typedef __attribute__((ext_vector_type(8)))  float  v8f;

union FragBF { v16bf v; uint4 q[2]; };

// ---------------------------------------------------------------------------
// fp32 -> bf16 (round to nearest even), 4 elements per thread
// ---------------------------------------------------------------------------
__device__ __forceinline__ unsigned f32_to_bf16(float f) {
  unsigned u = __float_as_uint(f);
  u += 0x7FFFu + ((u >> 16) & 1u);
  return u >> 16;
}

__global__ void cvt_f32_bf16_kernel(const float* __restrict__ src,
                                    unsigned short* __restrict__ dst, int n4) {
  int i = blockIdx.x * blockDim.x + threadIdx.x;
  if (i >= n4) return;
  float4 f = ((const float4*)src)[i];
  uint2 o;
  o.x = f32_to_bf16(f.x) | (f32_to_bf16(f.y) << 16);
  o.y = f32_to_bf16(f.z) | (f32_to_bf16(f.w) << 16);
  ((uint2*)dst)[i] = o;
}

// ---------------------------------------------------------------------------
// XW = X @ W_ih^T via v_wmma_f32_16x16x32_bf16.
// Block = 256 threads = 8 waves; each wave owns one 16x16 C tile.
// Block tile: 32(M) x 64(N).  Grid: (SEQ/32, HID/64).
// ---------------------------------------------------------------------------
__global__ __launch_bounds__(256) void gemm_xw_wmma(
    const unsigned short* __restrict__ Xb,   // [SEQ, HID] bf16 bits, row-major
    const unsigned short* __restrict__ Wb,   // [HID, HID] bf16 bits, row-major [n, k]
    float* __restrict__ XW)                  // [SEQ, HID] fp32
{
  const int lane = threadIdx.x & 31;
  const int wave = threadIdx.x >> 5;
  const int mBase = blockIdx.x * 32 + (wave & 1) * 16;
  const int nBase = blockIdx.y * 64 + (wave >> 1) * 16;

  const int l15 = lane & 15;
  const int g   = lane >> 4;

  const unsigned short* arow = Xb + (size_t)(mBase + l15) * HID;
  const unsigned short* brow = Wb + (size_t)(nBase + l15) * HID;

  v8f acc = {};
  #pragma unroll 4
  for (int k0 = 0; k0 < HID; k0 += 32) {
    FragBF a, b;
    // A fragment: lane (m = l15, g): K = [k0+8g, +8) and [k0+16+8g, +8)
    a.q[0] = *(const uint4*)(arow + k0 + 8 * g);
    a.q[1] = *(const uint4*)(arow + k0 + 16 + 8 * g);
    // B fragment (B = W^T): lane (n = l15, g): K = [k0+8g, +8) and [k0+16+8g, +8) of W row n
    b.q[0] = *(const uint4*)(brow + k0 + 8 * g);
    b.q[1] = *(const uint4*)(brow + k0 + 16 + 8 * g);

    __builtin_prefetch(arow + k0 + 256, 0, 1);   // global_prefetch_b8
    __builtin_prefetch(brow + k0 + 256, 0, 1);

    acc = __builtin_amdgcn_wmma_f32_16x16x32_bf16(
        /*neg_a=*/false, a.v, /*neg_b=*/false, b.v,
        /*c_mod=*/(short)0, acc, /*reuse_a=*/false, /*reuse_b=*/false);
  }

  // C layout: VGPR r -> row (r + 8*g), col l15
  const int m = mBase + g * 8;
  const int n = nBase + l15;
  #pragma unroll
  for (int r = 0; r < 8; ++r)
    XW[(size_t)(m + r) * HID + n] = acc[r];
}

// ---------------------------------------------------------------------------
// Persistent recurrent scan: 128 workgroups x 256 threads.
// Each workgroup holds 16 rows of W_hh (bf16) in 64KB LDS for the whole scan,
// filled once via the CDNA5 async global->LDS path (ASYNCcnt tracked).
// Per step: 16 dot-products of length 2048; 16 lanes per row, each owning
// K chunks {8*seg + 128*j} (16B LDS stride across segs -> conflict-free).
// Grid barrier per step via monotonic generation counter.
// ---------------------------------------------------------------------------
__global__ __launch_bounds__(256) void rnn_scan(
    const float* __restrict__ xw,            // [SEQ, HID]
    const unsigned short* __restrict__ whh,  // [HID, HID] bf16 bits
    float* __restrict__ ys,                  // [SEQ, HID]
    float* __restrict__ hbuf,                // [2, HID] ping-pong (pre-zeroed)
    unsigned* __restrict__ sync)             // [0]=arrive cnt, [1]=generation
{
  __shared__ unsigned short wsh[16 * HID];   // 64 KB

  const int r0 = blockIdx.x * 16;

  // ---- async fill of the LDS weight slice (global_load_async_to_lds_b128) ----
  {
    const unsigned lds_base = (unsigned)(uintptr_t)(&wsh[0]);  // low 32b = LDS offset
    const uint64_t gbase    = (uint64_t)(uintptr_t)(whh + (size_t)r0 * HID);
    // 64KB / 16B = 4096 b128 transfers; 16 per thread
    for (int i = threadIdx.x; i < (16 * HID * 2) / 16; i += blockDim.x) {
      unsigned laddr = lds_base + (unsigned)i * 16u;
      uint64_t gaddr = gbase + (uint64_t)i * 16u;
      asm volatile("global_load_async_to_lds_b128 %0, %1, off"
                   :: "v"(laddr), "v"(gaddr) : "memory");
    }
    asm volatile("s_wait_asynccnt 0" ::: "memory");
  }
  __syncthreads();

  const int row  = threadIdx.x >> 4;   // 0..15
  const int seg  = threadIdx.x & 15;   // 0..15
  const int gout = r0 + row;
  const uint4* wrow = (const uint4*)(wsh + row * HID) + seg;  // + j*16

  for (int t = 0; t < SEQ; ++t) {
    const float*  h  = hbuf + (t & 1) * HID;
    const float4* hp = (const float4*)h + 2 * seg;            // + j*32 (+1)

    float acc = 0.f;
    #pragma unroll
    for (int j = 0; j < 16; ++j) {
      uint4  wq = wrow[j * 16];        // 8 bf16 weights: K = 8*seg + 128*j ..+7
      float4 h0 = hp[j * 32];
      float4 h1 = hp[j * 32 + 1];
      acc = fmaf(__uint_as_float(wq.x << 16),          h0.x, acc);
      acc = fmaf(__uint_as_float(wq.x & 0xFFFF0000u),  h0.y, acc);
      acc = fmaf(__uint_as_float(wq.y << 16),          h0.z, acc);
      acc = fmaf(__uint_as_float(wq.y & 0xFFFF0000u),  h0.w, acc);
      acc = fmaf(__uint_as_float(wq.z << 16),          h1.x, acc);
      acc = fmaf(__uint_as_float(wq.z & 0xFFFF0000u),  h1.y, acc);
      acc = fmaf(__uint_as_float(wq.w << 16),          h1.z, acc);
      acc = fmaf(__uint_as_float(wq.w & 0xFFFF0000u),  h1.w, acc);
    }
    // reduce across the 16 lanes sharing this row (half-wave, wave32)
    #pragma unroll
    for (int off = 8; off >= 1; off >>= 1)
      acc += __shfl_xor(acc, off, 32);

    if (seg == 0) {
      float v = xw[(size_t)t * HID + gout] + acc;
      v = fmaxf(v, 0.f);
      ys[(size_t)t * HID + gout]          = v;
      hbuf[((t + 1) & 1) * HID + gout]    = v;
    }

    // -------- grid barrier (one per timestep) --------
    __threadfence();
    __syncthreads();
    if (threadIdx.x == 0) {
      unsigned a = atomicAdd(&sync[0], 1u);
      if (a == (unsigned)gridDim.x - 1u) {
        atomicExch(&sync[0], 0u);
        // release: publishes this step's h writes together with the gen bump
        __hip_atomic_fetch_add(&sync[1], 1u, __ATOMIC_RELEASE,
                               __HIP_MEMORY_SCOPE_AGENT);
      }
      // poll with a plain agent-scope load (no RMW traffic on L2 atomics)
      while (__hip_atomic_load(&sync[1], __ATOMIC_ACQUIRE,
                               __HIP_MEMORY_SCOPE_AGENT) < (unsigned)(t + 1))
        __builtin_amdgcn_s_sleep(1);
    }
    __syncthreads();
  }
}

__global__ void init_scan_state(float* hbuf, unsigned* sync) {
  int i = blockIdx.x * blockDim.x + threadIdx.x;
  if (i < 2 * HID) hbuf[i] = 0.f;
  if (i < 2) sync[i] = 0u;
}

__global__ void copy_hlast(const float* __restrict__ src, float* __restrict__ dst) {
  int i = blockIdx.x * blockDim.x + threadIdx.x;
  if (i < HID) dst[i] = src[i];
}

// ---------------------------------------------------------------------------
extern "C" void kernel_launch(void* const* d_in, const int* in_sizes, int n_in,
                              void* d_out, int out_size, void* d_ws, size_t ws_size,
                              hipStream_t stream) {
  (void)in_sizes; (void)n_in; (void)out_size; (void)ws_size;
  const float* x_in = (const float*)d_in[0];
  const float* w_ih = (const float*)d_in[1];
  const float* w_hh = (const float*)d_in[2];
  float* out = (float*)d_out;                 // [SEQ*HID] ys  + [HID] h_last

  char* ws = (char*)d_ws;
  size_t off = 0;
  auto alloc = [&](size_t bytes) -> char* {
    char* p = ws + off;
    off += (bytes + 255) & ~(size_t)255;
    return p;
  };
  unsigned short* xbf  = (unsigned short*)alloc((size_t)SEQ * HID * 2);  // 16 MB
  unsigned short* wihb = (unsigned short*)alloc((size_t)HID * HID * 2);  //  8 MB
  unsigned short* whhb = (unsigned short*)alloc((size_t)HID * HID * 2);  //  8 MB
  float*          xw   = (float*)alloc((size_t)SEQ * HID * 4);           // 32 MB
  float*          hbuf = (float*)alloc((size_t)2 * HID * 4);
  unsigned*       sync = (unsigned*)alloc(256);

  const int n4x = SEQ * HID / 4;
  const int n4w = HID * HID / 4;

  const float* x = x_in;
  for (int l = 0; l < NL; ++l) {
    cvt_f32_bf16_kernel<<<n4x / 256, 256, 0, stream>>>(x, xbf, n4x);
    cvt_f32_bf16_kernel<<<n4w / 256, 256, 0, stream>>>(w_ih + (size_t)l * HID * HID, wihb, n4w);
    cvt_f32_bf16_kernel<<<n4w / 256, 256, 0, stream>>>(w_hh + (size_t)l * HID * HID, whhb, n4w);

    dim3 g(SEQ / 32, HID / 64);
    gemm_xw_wmma<<<g, 256, 0, stream>>>(xbf, wihb, xw);

    init_scan_state<<<(2 * HID + 255) / 256, 256, 0, stream>>>(hbuf, sync);
    // ys of every layer goes to d_out (it is rewritten each layer; the last
    // layer's ys is the final output, which is what remains).
    rnn_scan<<<HID / 16, 256, 0, stream>>>(xw, whhb, out, hbuf, sync);
    x = out;
  }
  copy_hlast<<<(HID + 255) / 256, 256, 0, stream>>>(
      out + (size_t)(SEQ - 1) * HID, out + (size_t)SEQ * HID);
}
